// AsymmetricPositionAttentionModule_12180527251509
// MI455X (gfx1250) — matrix operand order repaired and checked by reading, not compiled, not run on hardware
//
#include <hip/hip_runtime.h>
#include <hip/hip_bf16.h>

// ---------------------------------------------------------------------------
// Types / helpers
// ---------------------------------------------------------------------------
typedef __attribute__((ext_vector_type(16))) __bf16 v16bf;
typedef __attribute__((ext_vector_type(8)))  float  v8f;

static __device__ __forceinline__ __bf16 to_bf(float f) {
  unsigned u = __float_as_uint(f);
  unsigned r = (u + 0x7FFFu + ((u >> 16) & 1u)) >> 16;  // RNE
  unsigned short s = (unsigned short)r;
  return __builtin_bit_cast(__bf16, s);
}
static __device__ __forceinline__ float to_f(__bf16 b) {
  unsigned short s = __builtin_bit_cast(unsigned short, b);
  return __uint_as_float(((unsigned)s) << 16);
}

// A-matrix (16x32 bf16) K index for fragment element e, lane-half `half`.
static __device__ __forceinline__ int a_kof(int e, int half) {
  int v = e >> 1;
  return ((v >> 2) << 4) + ((v & 3) << 1) + (e & 1) + (half << 3);
}
// B-matrix (32x16 bf16) K index: same order as A (lanes0-15 K 0..7,16..23; lanes16-31 K 8..15,24..31).
static __device__ __forceinline__ int b_kof(int e, int half) {
  int v = e >> 1;
  return ((v >> 2) << 4) + ((v & 3) << 1) + (e & 1) + (half << 3);
}

static __device__ __forceinline__ v8f wmma_bf16(const v16bf& a, const v16bf& b, const v8f& c) {
  return __builtin_amdgcn_wmma_f32_16x16x32_bf16(false, a, false, b, (short)0, c, false, false);
}

// CDNA5 async global->LDS copy (ASYNCcnt-tracked; 16B per lane).
// lds_off: byte offset into the workgroup's LDS; gaddr: 64-bit global address.
static __device__ __forceinline__ void async_copy16(unsigned lds_off, const void* gptr) {
  unsigned long long ga = (unsigned long long)(uintptr_t)gptr;
  asm volatile("global_load_async_to_lds_b128 %0, %1, off"
               :: "v"(lds_off), "v"(ga) : "memory");
}
static __device__ __forceinline__ void wait_async() {
  asm volatile("s_wait_asynccnt 0x0" ::: "memory");
}
static __device__ __forceinline__ unsigned lds_byte_off(const void* p) {
  return (unsigned)(uintptr_t)p;   // low 32 bits of generic ptr == LDS offset
}

// ---------------------------------------------------------------------------
// Format-conversion kernels
// ---------------------------------------------------------------------------
__global__ __launch_bounds__(256) void k_cvt_f32_bf16(const float* __restrict__ src,
                                                      __bf16* __restrict__ dst, int n) {
  int i = blockIdx.x * 256 + threadIdx.x;
  if (i < n) dst[i] = to_bf(src[i]);
}

// x [B,512,4096] fp32 (NCHW) -> xT [B*4096, 512] bf16
__global__ __launch_bounds__(256) void k_transpose_x(const float* __restrict__ x,
                                                     __bf16* __restrict__ xT) {
  int i = blockIdx.x * 256 + threadIdx.x;      // 0 .. 2^24-1
  int b   = i >> 21;
  int rem = i & ((1 << 21) - 1);
  int c   = rem >> 12;
  int m   = rem & 4095;
  xT[((size_t)(b * 4096 + m)) * 512 + c] = to_bf(x[i]);
}

__global__ __launch_bounds__(256) void k_fill0_bf16(__bf16* __restrict__ p, int n) {
  int i = blockIdx.x * 256 + threadIdx.x;
  if (i < n) p[i] = __builtin_bit_cast(__bf16, (unsigned short)0);
}

// ---------------------------------------------------------------------------
// 1x1 conv + BN + ReLU. Block = 8 waves; wave w owns m-tile (strip*8+w) and a
// 16x64 output strip (4 accumulators sharing one A fragment per k-step).
// Weight chunk (64 cols x 32 k, 4KB bf16) staged per k-step via async->LDS.
// A = xT [32768,512]; W = [Cout,512]; out bf16 [32768,Cout].
// ---------------------------------------------------------------------------
__global__ __launch_bounds__(256) void k_conv_bn_relu(
    const __bf16* __restrict__ aT, const __bf16* __restrict__ wq,
    const float* __restrict__ gamma, const float* __restrict__ beta,
    const float* __restrict__ mean,  const float* __restrict__ var,
    __bf16* __restrict__ outT, int ngroups) {
  __shared__ __bf16 tileB[64 * 32];            // [col][k], 4 KB
  const int Cin = 512;
  int t = threadIdx.x;
  int lane = t & 31, wid = t >> 5;
  int strip = blockIdx.x / ngroups;            // 0..255 (128-row strips)
  int ng    = blockIdx.x % ngroups;            // 64-col group
  int half = lane >> 4, l16 = lane & 15;
  int mrow = (strip * 8 + wid) * 16 + l16;
  const __bf16* arow = aT + (size_t)mrow * Cin;
  // this thread's async-copy slice of the weight chunk
  int ccol = t >> 2, cseg = t & 3;             // 64 cols x 4 x 16B
  unsigned ldst = lds_byte_off(tileB) + (unsigned)(ccol * 64 + cseg * 16);
  const __bf16* gsrc0 = wq + (size_t)(ng * 64 + ccol) * Cin + cseg * 8;

  v8f acc[4] = {};
  for (int kk = 0; kk < Cin; kk += 32) {
    __syncthreads();                           // previous chunk fully consumed
    async_copy16(ldst, gsrc0 + kk);
    wait_async();
    __syncthreads();                           // all 8 waves' slices landed
    v16bf af;
#pragma unroll
    for (int e = 0; e < 16; ++e) af[e] = arow[kk + a_kof(e, half)];
#pragma unroll
    for (int f = 0; f < 4; ++f) {
      v16bf bq;
#pragma unroll
      for (int e = 0; e < 16; ++e) bq[e] = tileB[(f * 16 + l16) * 32 + b_kof(e, half)];
      acc[f] = wmma_bf16(af, bq, acc[f]);
    }
  }
  int Cout = ngroups * 64;
#pragma unroll
  for (int f = 0; f < 4; ++f) {
    int c = ng * 64 + f * 16 + l16;
    float sc = gamma[c] * rsqrtf(var[c] + 1e-5f);
    float mu = mean[c], bt = beta[c];
#pragma unroll
    for (int v = 0; v < 8; ++v) {
      int m = (strip * 8 + wid) * 16 + v + half * 8;   // D row = v + half*8
      float y = fmaxf((acc[f][v] - mu) * sc + bt, 0.f);
      outT[(size_t)m * Cout + c] = to_bf(y);
    }
  }
}

// ---------------------------------------------------------------------------
// PSP adaptive max pool. One wave per (b, c, cell). cells: 1,3x3,6x6,8x8 = 110.
// mode 0: dst[b][cell][c]  ([B,112,C]);  mode 1: dst[b][c][cell] ([B,C,128]).
// ---------------------------------------------------------------------------
__global__ __launch_bounds__(256) void k_psp(const __bf16* __restrict__ srcT,
                                             __bf16* __restrict__ dst,
                                             int C, int mode, int totalWaves) {
  int lane = threadIdx.x & 31, wid = threadIdx.x >> 5;
  int wg = blockIdx.x * 8 + wid;
  if (wg >= totalWaves) return;
  int b = wg / (C * 110);
  int rem = wg % (C * 110);
  int c = rem / 110, cell = rem % 110;
  int s, i, j;
  if (cell == 0)       { s = 1; i = 0; j = 0; }
  else if (cell < 10)  { s = 3; int u = cell - 1;  i = u / 3; j = u % 3; }
  else if (cell < 46)  { s = 6; int u = cell - 10; i = u / 6; j = u % 6; }
  else                 { s = 8; int u = cell - 46; i = u / 8; j = u % 8; }
  int hs = (i * 64) / s, he = ((i + 1) * 64 + s - 1) / s;
  int ws = (j * 64) / s, we = ((j + 1) * 64 + s - 1) / s;
  int ww = we - ws, tot = (he - hs) * ww;
  float m = -1e30f;
  const __bf16* base = srcT + (size_t)b * 4096 * C + c;
  for (int u = lane; u < tot; u += 32) {
    int h = hs + u / ww, w = ws + u % ww;
    m = fmaxf(m, to_f(base[(size_t)(h * 64 + w) * C]));
  }
#pragma unroll
  for (int d = 1; d < 32; d <<= 1) m = fmaxf(m, __shfl_xor(m, d, 32));
  if (lane == 0) {
    if (mode == 0) dst[((size_t)b * 112 + cell) * C + c] = to_bf(m);
    else           dst[((size_t)b * C + c) * 128 + cell] = to_bf(m);
  }
}

// ---------------------------------------------------------------------------
// Fused attention: scores = (Q K)/16, softmax over 110, Y = P V.
// One wave per 16-row tile; 8 waves/block; each block covers one batch b.
// Whole key matrix for the batch (112x256 bf16 = 56KB) async-staged in LDS;
// after a phase barrier the same LDS buffer holds the probability tiles.
// ---------------------------------------------------------------------------
__global__ __launch_bounds__(256) void k_attn(const __bf16* __restrict__ qkT,   // [B*4096,256]
                                              const __bf16* __restrict__ keyT,  // [B,112,256]
                                              const __bf16* __restrict__ valP,  // [B,512,128]
                                              __bf16* __restrict__ yT) {        // [B*4096,512]
  __shared__ __bf16 smem[112 * 256];           // 56KB: key cache, then probs
  int t = threadIdx.x;
  int lane = t & 31, wid = t >> 5;
  int tile = blockIdx.x * 8 + wid;             // 0..2047, grid exact
  int b = tile >> 8;
  int m0 = (tile & 255) * 16;
  int half = lane >> 4, l16 = lane & 15;
  const int Ck = 256;

  // --- phase A: async-stage key[b] into LDS (14 rounds x 256 thr x 16B) ---
  {
    unsigned base = lds_byte_off(smem);
    const __bf16* gk = keyT + (size_t)b * 112 * 256;
#pragma unroll
    for (int r = 0; r < 14; ++r) {
      int idx = r * 256 + t;                   // 16B unit index
      async_copy16(base + (unsigned)idx * 16, gk + idx * 8);
    }
    wait_async();
  }
  __syncthreads();

  // --- phase B: scores GEMM M16 x K256 x N112 (7 n-tiles), B from LDS ---
  const __bf16* arow = qkT + (size_t)(b * 4096 + m0 + l16) * Ck;
  v8f scf[7] = {};
  for (int kk = 0; kk < Ck; kk += 32) {
    v16bf af;
#pragma unroll
    for (int e = 0; e < 16; ++e) af[e] = arow[kk + a_kof(e, half)];
#pragma unroll
    for (int f = 0; f < 7; ++f) {
      v16bf bq;
#pragma unroll
      for (int e = 0; e < 16; ++e)
        bq[e] = smem[(f * 16 + l16) * 256 + kk + b_kof(e, half)];
      scf[f] = wmma_bf16(af, bq, scf[f]);
    }
  }
  __syncthreads();                             // everyone done reading key

  // --- phase C: exact softmax; write P into this wave's LDS region ---
  __bf16* pr = smem + wid * (16 * 128);
#pragma unroll
  for (int v = 0; v < 8; ++v) {
    float vals[7];
    float mx = -1e30f;
#pragma unroll
    for (int f = 0; f < 7; ++f) {
      int s = f * 16 + l16;
      float x = (s < 110) ? scf[f][v] * 0.0625f : -1e30f;   // * Ck^-0.5
      vals[f] = x; mx = fmaxf(mx, x);
    }
#pragma unroll
    for (int d = 1; d < 16; d <<= 1) mx = fmaxf(mx, __shfl_xor(mx, d, 32));
    float sum = 0.f;
#pragma unroll
    for (int f = 0; f < 7; ++f) {
      int s = f * 16 + l16;
      float e = (s < 110) ? __expf(vals[f] - mx) : 0.f;
      vals[f] = e; sum += e;
    }
#pragma unroll
    for (int d = 1; d < 16; d <<= 1) sum += __shfl_xor(sum, d, 32);
    float r = 1.f / sum;
    int row = v + half * 8;
#pragma unroll
    for (int f = 0; f < 7; ++f)
      pr[row * 128 + f * 16 + l16] = to_bf(vals[f] * r);
    pr[row * 128 + 112 + l16] = __builtin_bit_cast(__bf16, (unsigned short)0);
  }
  __syncthreads();

  // --- phase D: Y = P V, M16 x K128 x N512 (32 n-tiles) ---
  const __bf16* prow = pr + l16 * 128;
  for (int nt = 0; nt < 32; ++nt) {
    v8f acc = {};
    const __bf16* brow = valP + ((size_t)b * 512 + nt * 16 + l16) * 128;
#pragma unroll
    for (int kk = 0; kk < 128; kk += 32) {
      v16bf af, bq;
#pragma unroll
      for (int e = 0; e < 16; ++e) {
        af[e] = prow[kk + a_kof(e, half)];
        bq[e] = brow[kk + b_kof(e, half)];
      }
      acc = wmma_bf16(af, bq, acc);
    }
#pragma unroll
    for (int v = 0; v < 8; ++v) {
      int m = m0 + v + half * 8;
      yT[((size_t)b * 4096 + m) * 512 + nt * 16 + l16] = to_bf(acc[v]);
    }
  }
}

// ---------------------------------------------------------------------------
// Output 1x1 conv + BN + residual; fp32 NCHW store. Same blocking as conv.
// ---------------------------------------------------------------------------
__global__ __launch_bounds__(256) void k_conv_out(
    const __bf16* __restrict__ yT, const __bf16* __restrict__ wq,
    const float* __restrict__ gamma, const float* __restrict__ beta,
    const float* __restrict__ mean,  const float* __restrict__ var,
    const float* __restrict__ xres, float* __restrict__ out) {
  __shared__ __bf16 tileB[64 * 32];
  const int Cin = 512, ngroups = 8;            // Cout = 512
  int t = threadIdx.x;
  int lane = t & 31, wid = t >> 5;
  int strip = blockIdx.x / ngroups;
  int ng    = blockIdx.x % ngroups;
  int half = lane >> 4, l16 = lane & 15;
  int mrow = (strip * 8 + wid) * 16 + l16;
  const __bf16* arow = yT + (size_t)mrow * Cin;
  int ccol = t >> 2, cseg = t & 3;
  unsigned ldst = lds_byte_off(tileB) + (unsigned)(ccol * 64 + cseg * 16);
  const __bf16* gsrc0 = wq + (size_t)(ng * 64 + ccol) * Cin + cseg * 8;

  v8f acc[4] = {};
  for (int kk = 0; kk < Cin; kk += 32) {
    __syncthreads();
    async_copy16(ldst, gsrc0 + kk);
    wait_async();
    __syncthreads();
    v16bf af;
#pragma unroll
    for (int e = 0; e < 16; ++e) af[e] = arow[kk + a_kof(e, half)];
#pragma unroll
    for (int f = 0; f < 4; ++f) {
      v16bf bq;
#pragma unroll
      for (int e = 0; e < 16; ++e) bq[e] = tileB[(f * 16 + l16) * 32 + b_kof(e, half)];
      acc[f] = wmma_bf16(af, bq, acc[f]);
    }
  }
#pragma unroll
  for (int f = 0; f < 4; ++f) {
    int c = ng * 64 + f * 16 + l16;
    float sc = gamma[c] * rsqrtf(var[c] + 1e-5f);
    float mu = mean[c], bt = beta[c];
#pragma unroll
    for (int v = 0; v < 8; ++v) {
      int mg = (strip * 8 + wid) * 16 + v + half * 8;
      int bb = mg >> 12, ml = mg & 4095;
      size_t idx = ((size_t)bb * 512 + c) * 4096 + ml;
      out[idx] = xres[idx] + (acc[f][v] - mu) * sc + bt;
    }
  }
}

// ---------------------------------------------------------------------------
// Launch
// ---------------------------------------------------------------------------
extern "C" void kernel_launch(void* const* d_in, const int* in_sizes, int n_in,
                              void* d_out, int out_size, void* d_ws, size_t ws_size,
                              hipStream_t stream) {
  const float* x        = (const float*)d_in[0];
  const float* qk_w     = (const float*)d_in[1];
  const float* qk_gamma = (const float*)d_in[2];
  const float* qk_beta  = (const float*)d_in[3];
  const float* qk_mean  = (const float*)d_in[4];
  const float* qk_var   = (const float*)d_in[5];
  const float* v_w      = (const float*)d_in[6];
  const float* v_gamma  = (const float*)d_in[7];
  const float* v_beta   = (const float*)d_in[8];
  const float* v_mean   = (const float*)d_in[9];
  const float* v_var    = (const float*)d_in[10];
  const float* out_w    = (const float*)d_in[11];
  const float* o_gamma  = (const float*)d_in[12];
  const float* o_beta   = (const float*)d_in[13];
  const float* o_mean   = (const float*)d_in[14];
  const float* o_var    = (const float*)d_in[15];
  float* out = (float*)d_out;

  // Workspace carve-up (bf16 elements)
  __bf16* ws = (__bf16*)d_ws;
  size_t o = 0;
  __bf16* xT   = ws + o; o += (size_t)8 * 4096 * 512;
  __bf16* qkT  = ws + o; o += (size_t)8 * 4096 * 256;
  __bf16* valT = ws + o; o += (size_t)8 * 4096 * 512;
  __bf16* yT   = ws + o; o += (size_t)8 * 4096 * 512;
  __bf16* keyT = ws + o; o += (size_t)8 * 112 * 256;    // padded S=112
  __bf16* valP = ws + o; o += (size_t)8 * 512 * 128;    // padded S=128
  __bf16* wqk  = ws + o; o += (size_t)256 * 512;
  __bf16* wv   = ws + o; o += (size_t)512 * 512;
  __bf16* wout = ws + o; o += (size_t)512 * 512;

  // 1) convert weights + transpose/convert x
  k_cvt_f32_bf16<<<(256 * 512 + 255) / 256, 256, 0, stream>>>(qk_w, wqk, 256 * 512);
  k_cvt_f32_bf16<<<(512 * 512 + 255) / 256, 256, 0, stream>>>(v_w, wv, 512 * 512);
  k_cvt_f32_bf16<<<(512 * 512 + 255) / 256, 256, 0, stream>>>(out_w, wout, 512 * 512);
  k_transpose_x<<<(1 << 24) / 256, 256, 0, stream>>>(x, xT);

  // 2) qk / value branches: 1x1 conv + BN + ReLU (WMMA + async weight staging)
  k_conv_bn_relu<<<256 * 4, 256, 0, stream>>>(xT, wqk, qk_gamma, qk_beta,
                                              qk_mean, qk_var, qkT, 4);
  k_conv_bn_relu<<<256 * 8, 256, 0, stream>>>(xT, wv, v_gamma, v_beta,
                                              v_mean, v_var, valT, 8);

  // 3) PSP pooling (zero-pad targets first so GEMM padding is inert)
  k_fill0_bf16<<<(8 * 112 * 256 + 255) / 256, 256, 0, stream>>>(keyT, 8 * 112 * 256);
  k_fill0_bf16<<<(8 * 512 * 128 + 255) / 256, 256, 0, stream>>>(valP, 8 * 512 * 128);
  {
    int waves_k = 8 * 256 * 110;
    k_psp<<<(waves_k + 7) / 8, 256, 0, stream>>>(qkT, keyT, 256, 0, waves_k);
    int waves_v = 8 * 512 * 110;
    k_psp<<<(waves_v + 7) / 8, 256, 0, stream>>>(valT, valP, 512, 1, waves_v);
  }

  // 4) fused scores -> softmax -> P*V (key async-staged in LDS)
  k_attn<<<2048 / 8, 256, 0, stream>>>(qkT, keyT, valP, yT);

  // 5) output conv + BN + residual
  k_conv_out<<<256 * 8, 256, 0, stream>>>(yT, wout, o_gamma, o_beta,
                                          o_mean, o_var, x, out);
}
